// FusedFuzzy_13572096655613
// MI455X (gfx1250) — matrix operand order repaired and checked by reading, not compiled, not run on hardware
//
#include <hip/hip_runtime.h>

typedef _Float16 h16;
typedef __attribute__((ext_vector_type(16))) _Float16 v16h;
typedef __attribute__((ext_vector_type(8)))  _Float16 v8h;
typedef __attribute__((ext_vector_type(8)))  float    v8f;

#define NFEAT 7
#define FUZZ  2187            // 3^7
#define FPAD  2208            // FUZZ padded to multiple of 32 (K of GEMM1)
#define JPAD  2240            // FUZZ padded to multiple of 64 (N of GEMM2)
#define MID   512
#define NCLS  10
#define BATCH 4096

// ---------------------------------------------------------------------------
// Kernel 1: Wsum[f,m,j] = sum_{k: digit_f(k)==m} wei[(k*7+f)*FUZZ + j]
// Reads wei exactly once (134 MB). k-split over blockIdx.z (27 chunks of 81),
// partials merged with f32 global atomics. Wsum pre-zeroed by hipMemsetAsync.
// ---------------------------------------------------------------------------
__global__ __launch_bounds__(256) void wsum_kernel(const float* __restrict__ wei,
                                                   float* __restrict__ Wsum) {
  int j = blockIdx.x * 256 + threadIdx.x;
  if (j >= FUZZ) return;
  int f = blockIdx.y;
  int kStart = blockIdx.z * 81;
  const int p3[7] = {729, 243, 81, 27, 9, 3, 1};
  int pf = p3[f];
  float s0 = 0.f, s1 = 0.f, s2 = 0.f;
  for (int kk = 0; kk < 81; ++kk) {
    int k = kStart + kk;
    int d = (k / pf) % 3;
    float w = wei[((size_t)(k * 7 + f)) * FUZZ + j];
    if (d == 0) s0 += w; else if (d == 1) s1 += w; else s2 += w;
  }
  atomicAdd(&Wsum[(f * 3 + 0) * FPAD + j], s0);
  atomicAdd(&Wsum[(f * 3 + 1) * FPAD + j], s1);
  atomicAdd(&Wsum[(f * 3 + 2) * FPAD + j], s2);
}

// ---------------------------------------------------------------------------
// Weight packs to f16 with zero padding.
// ---------------------------------------------------------------------------
__global__ __launch_bounds__(256) void packW1_kernel(const float* __restrict__ W1,
                                                     h16* __restrict__ W1h) {
  int idx = blockIdx.x * 256 + threadIdx.x;           // [MID][FPAD]
  if (idx >= MID * FPAD) return;
  int o = idx / FPAD, jc = idx % FPAD;
  W1h[idx] = (jc < FUZZ) ? (h16)W1[(size_t)o * FUZZ + jc] : (h16)0.f;
}

__global__ __launch_bounds__(256) void packW2_kernel(const float* __restrict__ W2,
                                                     h16* __restrict__ W2h) {
  int idx = blockIdx.x * 256 + threadIdx.x;           // [JPAD][MID]
  if (idx >= JPAD * MID) return;
  int j = idx / MID, o = idx % MID;
  W2h[idx] = (j < FUZZ) ? (h16)W2[(size_t)j * MID + o] : (h16)0.f;
}

// ---------------------------------------------------------------------------
// Kernel 4: a_mean. One block per batch row.
//   u[f,m] = exp(-(x-c)^2/b^2); a[j] = bais + sum_{21} u*Wsum[:,j];
//   a_mean[j] = a[j]/sum_j a[j]; stored f16, columns [FUZZ,FPAD) zeroed.
// ---------------------------------------------------------------------------
__global__ __launch_bounds__(256) void amean_kernel(const float* __restrict__ x,
                                                    const float* __restrict__ c,
                                                    const float* __restrict__ bb,
                                                    const float* __restrict__ bais,
                                                    const float* __restrict__ Wsum,
                                                    h16* __restrict__ Ah) {
  __shared__ float uS[21];
  __shared__ float aS[FUZZ];
  __shared__ float red[256];
  int b = blockIdx.x;
  int tid = threadIdx.x;
  if (tid < 21) {
    int f = tid / 3, m = tid % 3;
    float d  = x[b * NFEAT + f] - c[f * 3 + m];
    float bv = bb[f * 3 + m];
    uS[tid] = expf(-(d * d) / (bv * bv));
  }
  __syncthreads();
  float u[21];
#pragma unroll
  for (int t = 0; t < 21; ++t) u[t] = uS[t];
  float bias = bais[0];
  float lsum = 0.f;
  for (int j = tid; j < FUZZ; j += 256) {
    float a = bias;
#pragma unroll
    for (int t = 0; t < 21; ++t) a += u[t] * Wsum[t * FPAD + j];
    aS[j] = a;
    lsum += a;
  }
  red[tid] = lsum;
  __syncthreads();
  for (int s = 128; s > 0; s >>= 1) {
    if (tid < s) red[tid] += red[tid + s];
    __syncthreads();
  }
  float inv = 1.0f / red[0];
  for (int j = tid; j < FPAD; j += 256) {
    float v = (j < FUZZ) ? aS[j] * inv : 0.f;
    Ah[(size_t)b * FPAD + j] = (h16)v;
  }
}

// ---------------------------------------------------------------------------
// WMMA fragment loaders (16-bit, per CDNA5 ISA 7.12.2 layouts).
// A 16x32 (MxK): lane m=L%16, half=L/16; elems 0..7 -> K=half*8+e,
//                elems 8..15 -> K=16+half*8+(e-8).
// B 32x16 (KxN): lane n=L%16, half=L/16; same K mapping as A.
// ---------------------------------------------------------------------------
__device__ inline v16h load_a_frag(const h16* __restrict__ A, int rowStride,
                                   int m0, int k0, int lane) {
  int m = m0 + (lane & 15);
  int half = lane >> 4;
  const h16* p = A + (size_t)m * rowStride + k0 + half * 8;
  v8h lo = *(const v8h*)(p);
  v8h hi = *(const v8h*)(p + 16);
  v16h r;
#pragma unroll
  for (int i = 0; i < 8; ++i) { r[i] = lo[i]; r[i + 8] = hi[i]; }
  return r;
}

__device__ inline v16h load_b_frag(const h16* __restrict__ B, int rowStride,
                                   int n0, int k0, int lane) {
  int n = n0 + (lane & 15);
  int half = lane >> 4;
  const h16* p = B + (size_t)n * rowStride + k0 + half * 8;
  v8h lo = *(const v8h*)(p);
  v8h hi = *(const v8h*)(p + 16);
  v16h r;
#pragma unroll
  for (int i = 0; i < 8; ++i) { r[i] = lo[i]; r[i + 8] = hi[i]; }
  return r;
}

// ---------------------------------------------------------------------------
// GEMM1: H1[b,o] = sum_j Ah[b,j]*W1h[o,j] + b1[o]   (4096x2208 @ 2208x512)
// One wave computes a 16(M)x64(N) strip; block = 4 waves stacked in M.
// ---------------------------------------------------------------------------
__global__ __launch_bounds__(128) void gemm1_kernel(const h16* __restrict__ Ah,
                                                    const h16* __restrict__ W1h,
                                                    const float* __restrict__ b1,
                                                    h16* __restrict__ H1h) {
  int lane = threadIdx.x & 31;
  int wave = threadIdx.x >> 5;
  int m0 = (blockIdx.y * 4 + wave) * 16;
  int n0 = blockIdx.x * 64;
  v8f acc[4] = {v8f{}, v8f{}, v8f{}, v8f{}};
  for (int k0 = 0; k0 < FPAD; k0 += 32) {
    v16h a = load_a_frag(Ah, FPAD, m0, k0, lane);
#pragma unroll
    for (int sub = 0; sub < 4; ++sub) {
      v16h bfr = load_b_frag(W1h, FPAD, n0 + sub * 16, k0, lane);
      acc[sub] = __builtin_amdgcn_wmma_f32_16x16x32_f16(
          false, a, false, bfr, (short)0, acc[sub], false, false);
    }
  }
  int mRow = (lane >> 4) << 3;
  int nCol = lane & 15;
#pragma unroll
  for (int sub = 0; sub < 4; ++sub) {
    int o = n0 + sub * 16 + nCol;
    float bias = b1[o];
#pragma unroll
    for (int r = 0; r < 8; ++r) {
      int brow = m0 + mRow + r;
      H1h[(size_t)brow * MID + o] = (h16)(acc[sub][r] + bias);
    }
  }
}

// ---------------------------------------------------------------------------
// GEMM2: S[b,j] = a_mean[b,j] + b2[j] + sum_o H1[b,o]*W2h[j,o]
// (4096x512 @ 512x2240); residual + bias fused in epilogue, stored f16.
// ---------------------------------------------------------------------------
__global__ __launch_bounds__(128) void gemm2_kernel(const h16* __restrict__ H1h,
                                                    const h16* __restrict__ W2h,
                                                    const float* __restrict__ b2,
                                                    const h16* __restrict__ Ah,
                                                    h16* __restrict__ Sh) {
  int lane = threadIdx.x & 31;
  int wave = threadIdx.x >> 5;
  int m0 = (blockIdx.y * 4 + wave) * 16;
  int n0 = blockIdx.x * 64;
  v8f acc[4] = {v8f{}, v8f{}, v8f{}, v8f{}};
  for (int k0 = 0; k0 < MID; k0 += 32) {
    v16h a = load_a_frag(H1h, MID, m0, k0, lane);
#pragma unroll
    for (int sub = 0; sub < 4; ++sub) {
      v16h bfr = load_b_frag(W2h, MID, n0 + sub * 16, k0, lane);
      acc[sub] = __builtin_amdgcn_wmma_f32_16x16x32_f16(
          false, a, false, bfr, (short)0, acc[sub], false, false);
    }
  }
  int mRow = (lane >> 4) << 3;
  int nCol = lane & 15;
#pragma unroll
  for (int sub = 0; sub < 4; ++sub) {
    int jG = n0 + sub * 16 + nCol;
    float bias = (jG < FUZZ) ? b2[jG] : 0.f;
#pragma unroll
    for (int r = 0; r < 8; ++r) {
      int brow = m0 + mRow + r;
      float am = (jG < FUZZ) ? (float)Ah[(size_t)brow * FPAD + jG] : 0.f;
      Sh[(size_t)brow * JPAD + jG] = (h16)(acc[sub][r] + bias + am);
    }
  }
}

// ---------------------------------------------------------------------------
// Kernel 7: out[b,c] = leaky_relu( b3[c] + sum_j S[b,j]*W3[c,j], 0.2 )
// One block per batch row; wave32 shuffle + LDS reduction.
// ---------------------------------------------------------------------------
__global__ __launch_bounds__(256) void gemm3_kernel(const h16* __restrict__ Sh,
                                                    const float* __restrict__ W3,
                                                    const float* __restrict__ b3,
                                                    float* __restrict__ out) {
  int b = blockIdx.x, tid = threadIdx.x;
  const h16* srow = Sh + (size_t)b * JPAD;
  float acc[NCLS];
#pragma unroll
  for (int c = 0; c < NCLS; ++c) acc[c] = 0.f;
  for (int j = tid; j < FUZZ; j += 256) {
    float s = (float)srow[j];
#pragma unroll
    for (int c = 0; c < NCLS; ++c) acc[c] += s * W3[(size_t)c * FUZZ + j];
  }
  __shared__ float red[NCLS][8];
  int lane = tid & 31, wave = tid >> 5;
#pragma unroll
  for (int c = 0; c < NCLS; ++c) {
    float v = acc[c];
    for (int off = 16; off > 0; off >>= 1) v += __shfl_down(v, off, 32);
    if (lane == 0) red[c][wave] = v;
  }
  __syncthreads();
  if (tid < NCLS) {
    float t = 0.f;
#pragma unroll
    for (int w = 0; w < 8; ++w) t += red[tid][w];
    t += b3[tid];
    out[(size_t)b * NCLS + tid] = (t >= 0.f) ? t : 0.2f * t;
  }
}

// ---------------------------------------------------------------------------
extern "C" void kernel_launch(void* const* d_in, const int* in_sizes, int n_in,
                              void* d_out, int out_size, void* d_ws, size_t ws_size,
                              hipStream_t stream) {
  (void)in_sizes; (void)n_in; (void)out_size; (void)ws_size;
  const float* x    = (const float*)d_in[0];
  const float* c    = (const float*)d_in[1];
  const float* bb   = (const float*)d_in[2];
  const float* wei  = (const float*)d_in[3];
  const float* bais = (const float*)d_in[4];
  const float* W1   = (const float*)d_in[5];
  const float* b1   = (const float*)d_in[6];
  const float* W2   = (const float*)d_in[7];
  const float* b2   = (const float*)d_in[8];
  const float* W3   = (const float*)d_in[9];
  const float* b3   = (const float*)d_in[10];
  float* out = (float*)d_out;

  char* ws = (char*)d_ws;
  size_t off = 0;
  auto alloc = [&](size_t bytes) -> void* {
    void* p = ws + off;
    off += (bytes + 255) & ~(size_t)255;
    return p;
  };
  float* Wsum = (float*)alloc((size_t)21 * FPAD * sizeof(float));   // ~0.2 MB
  h16*   W1h  = (h16*)alloc((size_t)MID * FPAD * sizeof(h16));      // 2.3 MB
  h16*   W2h  = (h16*)alloc((size_t)JPAD * MID * sizeof(h16));      // 2.3 MB
  h16*   Ah   = (h16*)alloc((size_t)BATCH * FPAD * sizeof(h16));    // 18.1 MB
  h16*   H1h  = (h16*)alloc((size_t)BATCH * MID * sizeof(h16));     // 4.2 MB
  h16*   Sh   = (h16*)alloc((size_t)BATCH * JPAD * sizeof(h16));    // 18.4 MB

  hipMemsetAsync(Wsum, 0, (size_t)21 * FPAD * sizeof(float), stream);
  wsum_kernel<<<dim3(9, 7, 27), 256, 0, stream>>>(wei, Wsum);
  packW1_kernel<<<(MID * FPAD + 255) / 256, 256, 0, stream>>>(W1, W1h);
  packW2_kernel<<<(JPAD * MID + 255) / 256, 256, 0, stream>>>(W2, W2h);
  amean_kernel<<<BATCH, 256, 0, stream>>>(x, c, bb, bais, Wsum, Ah);
  gemm1_kernel<<<dim3(MID / 64, BATCH / 64), 128, 0, stream>>>(Ah, W1h, b1, H1h);
  gemm2_kernel<<<dim3(JPAD / 64, BATCH / 64), 128, 0, stream>>>(H1h, W2h, b2, Ah, Sh);
  gemm3_kernel<<<BATCH, 256, 0, stream>>>(Sh, W3, b3, out);
}
